// EdgeGCN_41437844472001
// MI455X (gfx1250) — matrix operand order repaired and checked by reading, not compiled, not run on hardware
//
#include <hip/hip_runtime.h>
#include <math.h>

typedef unsigned int   u32;
typedef unsigned short u16;
typedef __attribute__((ext_vector_type(2)))  u32    u32x2;
typedef __attribute__((ext_vector_type(4)))  u32    u32x4;
typedef __attribute__((ext_vector_type(4)))  float  f32x4;
typedef __attribute__((ext_vector_type(16))) __bf16 v16bf;
typedef __attribute__((ext_vector_type(8)))  float  v8f;

#define GLO __attribute__((address_space(1)))   // force global_load/store lowering

#define N_NODES_C  50000
#define N_EDGES_C  800000
#define D_FEAT_C   128
#define D_MODEL_C  256
#define TM_C       128                     // edges per workgroup
#define LDS_COLS_C 264                     // 256 + 8 bf16 pad -> 528B row stride
#define OUT_ELEMS  (N_NODES_C * D_MODEL_C)

__device__ __forceinline__ u16 f2bf(float f) {
  u32 u = __float_as_uint(f);
  u32 r = u + 0x7fffu + ((u >> 16) & 1u);   // round-to-nearest-even
  return (u16)(r >> 16);
}

union Frag { u32x4 q[2]; v16bf v; };

// ---------------------------------------------------------------- init / fixup
__global__ void init_out_kernel(float* __restrict__ out) {
  int i = blockIdx.x * 256 + threadIdx.x;
  if (i < OUT_ELEMS) ((GLO float*)out)[i] = __int_as_float(0xff800000); // -inf
}

__global__ void fixup_out_kernel(float* __restrict__ out) {
  int i = blockIdx.x * 256 + threadIdx.x;
  if (i < OUT_ELEMS) {
    float v = ((const GLO float*)out)[i];
    ((GLO float*)out)[i] = __builtin_isfinite(v) ? v : 0.0f;
  }
}

// ------------------------------------------------- weight transpose+cvt (bf16)
// Wt layout: [layer][n][k], k contiguous (so B-frags are contiguous b128 loads)
__global__ void prep_weights_kernel(const float* __restrict__ W1,
                                    const float* __restrict__ W2,
                                    const float* __restrict__ W3,
                                    u16* __restrict__ Wt) {
  int i = blockIdx.x * 256 + threadIdx.x;   // 3*256*256
  if (i >= 3 * 65536) return;
  int l   = i >> 16;
  int rem = i & 65535;
  int n   = rem >> 8;
  int k   = rem & 255;
  const float* W = (l == 0) ? W1 : (l == 1) ? W2 : W3;
  ((GLO u16*)Wt)[i] = f2bf(((const GLO float*)W)[k * D_MODEL_C + n]);
}

// ------------------------------------------------------------- fused EdgeConv
__global__ __launch_bounds__(256, 1) void edgeconv_fused_kernel(
    const float* __restrict__ x, const int* __restrict__ ei,
    const u16* __restrict__ Wt, const float* __restrict__ bias1,
    const float* __restrict__ bias2, const float* __restrict__ bias3,
    float* __restrict__ out) {
  __shared__ u16 Abuf[TM_C * LDS_COLS_C];   // ~67.6 KB (gfx1250: 320 KB/WGP)
  __shared__ int dstIdx[TM_C];

  const int tid = threadIdx.x;
  const int e0  = blockIdx.x * TM_C;

  // ---- gather: row e = [bf16(x_i) | bf16(x_j - x_i)], 2 threads per edge ----
  {
    const int le   = tid >> 1;      // local edge 0..127
    const int half = tid & 1;       // feature half
    const int e    = e0 + le;
    const int src  = ((const GLO int*)ei)[e];
    const int dst  = ((const GLO int*)ei)[N_EDGES_C + e];
    if (half == 0) dstIdx[le] = dst;
    const GLO f32x4* xi = (const GLO f32x4*)(x + (long)dst * D_FEAT_C);
    const GLO f32x4* xj = (const GLO f32x4*)(x + (long)src * D_FEAT_C);
    u16* rowp = Abuf + le * LDS_COLS_C;
#pragma unroll
    for (int q = 0; q < 16; ++q) {
      const int f4 = half * 16 + q;
      f32x4 a = xi[f4];
      f32x4 b = xj[f4];
      u32x2 pa, pd;
      pa.x = (u32)f2bf(a.x) | ((u32)f2bf(a.y) << 16);
      pa.y = (u32)f2bf(a.z) | ((u32)f2bf(a.w) << 16);
      pd.x = (u32)f2bf(b.x - a.x) | ((u32)f2bf(b.y - a.y) << 16);
      pd.y = (u32)f2bf(b.z - a.z) | ((u32)f2bf(b.w - a.w) << 16);
      *(u32x2*)(rowp + f4 * 4) = pa;
      *(u32x2*)(rowp + D_FEAT_C + f4 * 4) = pd;
    }
  }
  __syncthreads();

  const int lane = tid & 31;
  const int wave = tid >> 5;   // 8 waves; wave owns output columns [32w, 32w+32)
  const int l15  = lane & 15;
  const int lhi  = lane >> 4;

  const GLO u16* WtG = (const GLO u16*)Wt;
  const GLO u16* WtL[3] = {WtG, WtG + 65536, WtG + 2 * 65536};
  const GLO float* bL[3] = {(const GLO float*)bias1, (const GLO float*)bias2,
                            (const GLO float*)bias3};

#pragma unroll 1
  for (int l = 0; l < 3; ++l) {
    const GLO u16*   Wl = WtL[l];
    const GLO float* bv = bL[l];
    const float bs0 = bv[wave * 32 + l15];
    const float bs1 = bv[wave * 32 + 16 + l15];

    v8f acc[2][8];
#pragma unroll
    for (int t = 0; t < 2; ++t)
#pragma unroll
      for (int m = 0; m < 8; ++m)
        acc[t][m] = (v8f){0.f, 0.f, 0.f, 0.f, 0.f, 0.f, 0.f, 0.f};

#pragma unroll 1
    for (int ks = 0; ks < 8; ++ks) {
      // B-frags for 2 N-tiles: 16 contiguous K values per lane (Wt is n-major)
      Frag fb0, fb1;
      const GLO u16* p0 = Wl + (wave * 32 + l15) * D_MODEL_C + ks * 32 + lhi * 16;
      const GLO u16* p1 = p0 + 16 * D_MODEL_C;
      fb0.q[0] = ((const GLO u32x4*)p0)[0];
      fb0.q[1] = ((const GLO u32x4*)p0)[1];
      fb1.q[0] = ((const GLO u32x4*)p1)[0];
      fb1.q[1] = ((const GLO u32x4*)p1)[1];
#pragma unroll
      for (int m = 0; m < 8; ++m) {
        // A-frag from LDS: two b128 chunks, 16 contiguous K values per lane (matches B)
        Frag fa;
        const u16* ap = Abuf + (m * 16 + l15) * LDS_COLS_C + ks * 32 + lhi * 16;
        fa.q[0] = ((const u32x4*)ap)[0];
        fa.q[1] = ((const u32x4*)(ap + 8))[0];
        acc[0][m] = __builtin_amdgcn_wmma_f32_16x16x32_bf16(
            false, fa.v, false, fb0.v, (short)0, acc[0][m], false, false);
        acc[1][m] = __builtin_amdgcn_wmma_f32_16x16x32_bf16(
            false, fa.v, false, fb1.v, (short)0, acc[1][m], false, false);
      }
    }

    if (l < 2) {
      __syncthreads();   // everyone done reading Abuf for this layer
#pragma unroll
      for (int m = 0; m < 8; ++m) {
#pragma unroll
        for (int r = 0; r < 8; ++r) {
          const int row = m * 16 + r + 8 * lhi;
          float v0 = fmaxf(acc[0][m][r] + bs0, 0.f);
          float v1 = fmaxf(acc[1][m][r] + bs1, 0.f);
          Abuf[row * LDS_COLS_C + wave * 32 + l15]      = f2bf(v0);
          Abuf[row * LDS_COLS_C + wave * 32 + 16 + l15] = f2bf(v1);
        }
      }
      __syncthreads();
    } else {
      // final layer: bias + scatter-max via native f32 max atomic
#pragma unroll
      for (int m = 0; m < 8; ++m) {
#pragma unroll
        for (int r = 0; r < 8; ++r) {
          const int row = m * 16 + r + 8 * lhi;
          const int d   = dstIdx[row];
          float v0 = acc[0][m][r] + bs0;
          float v1 = acc[1][m][r] + bs1;
          unsigned long long a0 =
              (unsigned long long)(out + (long)d * D_MODEL_C + wave * 32 + l15);
          unsigned long long a1 = a0 + 16ull * 4ull;
          asm volatile("global_atomic_max_num_f32 %0, %1, off scope:SCOPE_DEV"
                       :: "v"(a0), "v"(v0) : "memory");
          asm volatile("global_atomic_max_num_f32 %0, %1, off scope:SCOPE_DEV"
                       :: "v"(a1), "v"(v1) : "memory");
        }
      }
    }
  }
}

// -------------------------------------------------------------------- launch
extern "C" void kernel_launch(void* const* d_in, const int* in_sizes, int n_in,
                              void* d_out, int out_size, void* d_ws, size_t ws_size,
                              hipStream_t stream) {
  const float* x   = (const float*)d_in[0];
  const int*   ei  = (const int*)d_in[1];
  const float* W1  = (const float*)d_in[2];
  const float* b1v = (const float*)d_in[3];
  const float* W2  = (const float*)d_in[4];
  const float* b2v = (const float*)d_in[5];
  const float* W3  = (const float*)d_in[6];
  const float* b3v = (const float*)d_in[7];
  float* out = (float*)d_out;
  u16*   Wt  = (u16*)d_ws;   // 3 * 256 * 256 * 2B = 384 KB

  init_out_kernel<<<(OUT_ELEMS + 255) / 256, 256, 0, stream>>>(out);
  prep_weights_kernel<<<(3 * 65536 + 255) / 256, 256, 0, stream>>>(W1, W2, W3, Wt);
  edgeconv_fused_kernel<<<N_EDGES_C / TM_C, 256, 0, stream>>>(
      x, ei, Wt, b1v, b2v, b3v, out);
  fixup_out_kernel<<<(OUT_ELEMS + 255) / 256, 256, 0, stream>>>(out);
}
